// update_v_79044578116056
// MI455X (gfx1250) — compile-verified, matrix operand order checked by reading, not run on hardware
//
#include <hip/hip_runtime.h>

// ---------------------------------------------------------------------------
// CDNA5 (gfx1250) implementation of the "update_v" GNN block.
// Matmuls use v_wmma_f32_16x16x32_bf16; segment-sums use global f32 atomics.
// ---------------------------------------------------------------------------

typedef __attribute__((ext_vector_type(16))) __bf16 v16bf;
typedef __attribute__((ext_vector_type(8)))  __bf16 v8bf;
typedef __attribute__((ext_vector_type(8)))  float  v8f;

#define N_NODES 100000
#define E_MAIN  800000
#define E_HULL  400000

__device__ __forceinline__ __bf16 f2bf(float f) {
  unsigned u = __builtin_bit_cast(unsigned, f);
  unsigned r = u + 0x7FFFu + ((u >> 16) & 1u);   // round-to-nearest-even
  unsigned short h = (unsigned short)(r >> 16);
  return __builtin_bit_cast(__bf16, h);
}

__device__ __forceinline__ float silu_f(float x) { return x / (1.0f + __expf(-x)); }

// Load one 16x32 bf16 WMMA fragment (A layout; B uses the same pattern on the
// pre-transposed [Nout, K] weight). Row-major source, stride `ld` elements.
// Lanes 0-15: elems 0..7 -> K=k0..k0+7, elems 8..15 -> K=k0+16..k0+23.
// Lanes 16-31: both ranges shifted by +8.  Each half is one contiguous B128.
__device__ __forceinline__ v16bf load_frag(const __bf16* __restrict__ base,
                                           int ld, int row0, int k0, int lane) {
  const __bf16* p = base + (size_t)(row0 + (lane & 15)) * ld + k0 + ((lane & 16) >> 1);
  v16bf f;
#pragma unroll
  for (int e = 0; e < 8; ++e) f[e] = p[e];
#pragma unroll
  for (int e = 0; e < 8; ++e) f[8 + e] = p[16 + e];
  return f;
}

// ---------------------------------------------------------------------------
// Utility kernels
// ---------------------------------------------------------------------------
__global__ void zero_f32(float* __restrict__ p, size_t n) {
  size_t t = (size_t)blockIdx.x * blockDim.x + threadIdx.x;
  if (t < n) p[t] = 0.0f;
}

// v = segment_sum(e2, i)  -- one thread per (edge, 4 channels), float4 loads
__global__ void scatter_v_kernel(const float4* __restrict__ e2v,
                                 const int* __restrict__ idx,
                                 float* __restrict__ v) {
  size_t t = (size_t)blockIdx.x * blockDim.x + threadIdx.x;
  if (t >= (size_t)E_MAIN * 32) return;
  int e  = (int)(t >> 5);
  int c4 = (int)(t & 31) << 2;
  float4 d = e2v[t];
  float* dst = v + (size_t)idx[e] * 128 + c4;
  atomicAdd(dst + 0, d.x);
  atomicAdd(dst + 1, d.y);
  atomicAdd(dst + 2, d.z);
  atomicAdd(dst + 3, d.w);
}

// f32 [M,C] contiguous -> bf16 at dst[row*ldd + coloff + col]
__global__ void f32_to_bf16_kernel(const float* __restrict__ src,
                                   __bf16* __restrict__ dst,
                                   int M, int C, int ldd, int coloff) {
  size_t t = (size_t)blockIdx.x * blockDim.x + threadIdx.x;
  if (t >= (size_t)M * C) return;
  int r = (int)(t / C), c = (int)(t % C);
  dst[(size_t)r * ldd + coloff + c] = f2bf(src[t]);
}

// transpose + convert weight [in,out] f32 -> [out, in_pad] bf16 (zero padded K)
__global__ void prep_wt_kernel(const float* __restrict__ w, __bf16* __restrict__ wt,
                               int in_dim, int out_dim, int in_pad) {
  size_t t = (size_t)blockIdx.x * blockDim.x + threadIdx.x;
  if (t >= (size_t)out_dim * in_pad) return;
  int o = (int)(t / in_pad), k = (int)(t % in_pad);
  float val = (k < in_dim) ? w[(size_t)k * out_dim + o] : 0.0f;
  wt[t] = f2bf(val);
}

// fea_hull [EH,16] f32 -> [EH,32] bf16 zero-padded (so layer-1 K=32 is clean)
__global__ void prep_fea_kernel(const float* __restrict__ fea, __bf16* __restrict__ dst) {
  size_t t = (size_t)blockIdx.x * blockDim.x + threadIdx.x;
  if (t >= (size_t)E_HULL * 32) return;
  int e = (int)(t >> 5), k = (int)(t & 31);
  float val = (k < 16) ? fea[(size_t)e * 16 + k] : 0.0f;
  dst[t] = f2bf(val);
}

// ---------------------------------------------------------------------------
// Generic WMMA GEMM: C[M,Nout] = act(A[M,K]_bf16 @ Wt[Nout,K]^T + bias)
// One wave per 16x64 output tile (4 accumulators share one A fragment).
// M multiple of 16; Nout multiple of 64; K multiple of 32.
// ---------------------------------------------------------------------------
__global__ void gemm_bf16_kernel(const __bf16* __restrict__ A, int lda,
                                 const __bf16* __restrict__ Wt,  // [Nout, K]
                                 const float* __restrict__ bias, // may be null
                                 int M, int K, int Nout,
                                 float* __restrict__ Cf,         // f32 out (or null)
                                 __bf16* __restrict__ Cb,        // bf16 out (or null)
                                 int ldc, int coloff, int do_silu) {
  int wave = blockIdx.x * (blockDim.x >> 5) + (threadIdx.x >> 5);
  int lane = threadIdx.x & 31;
  int ngrp = Nout >> 6;                 // groups of 4 n-tiles
  int tm = wave / ngrp;
  int tg = wave - tm * ngrp;
  if ((tm << 4) >= M) return;
  int m0 = tm << 4, n0 = tg << 6;

  v8f acc[4];
#pragma unroll
  for (int q = 0; q < 4; ++q) {
    v8f z = {0.f, 0.f, 0.f, 0.f, 0.f, 0.f, 0.f, 0.f};
    acc[q] = z;
  }
  for (int k0 = 0; k0 < K; k0 += 32) {
    v16bf a = load_frag(A, lda, m0, k0, lane);
#pragma unroll
    for (int q = 0; q < 4; ++q) {
      v16bf b = load_frag(Wt, K, n0 + (q << 4), k0, lane);
      acc[q] = __builtin_amdgcn_wmma_f32_16x16x32_bf16(false, a, false, b,
                                                       (short)0, acc[q], false, false);
    }
  }

  int nl = lane & 15;
  int mbase = m0 + ((lane & 16) >> 1);   // +8 for lanes 16-31
#pragma unroll
  for (int q = 0; q < 4; ++q) {
    int n = n0 + (q << 4) + nl;
    float bv = bias ? bias[n] : 0.0f;
#pragma unroll
    for (int r = 0; r < 8; ++r) {
      float x = acc[q][r] + bv;
      if (do_silu) x = silu_f(x);
      size_t off = (size_t)(mbase + r) * ldc + coloff + n;
      if (Cb) Cb[off] = f2bf(x);
      else    Cf[off] = x;
    }
  }
}

// ---------------------------------------------------------------------------
// Fused edge kernel:
//   W_h = silu(feaP @ w0t^T + b0) @ w1t^T + b1          (two WMMA layers)
//   e_hull = v_hull[j] * W_h ; out_hull[i] += e_hull    (gather-mul-scatter)
// One wave per 16-edge tile. LDS stages the 16x128 bf16 hidden activations.
// ---------------------------------------------------------------------------
__global__ void edge_mlp_scatter_kernel(const __bf16* __restrict__ feaP,  // [EH,32]
                                        const __bf16* __restrict__ w0t,   // [128,32]
                                        const float*  __restrict__ b0,
                                        const __bf16* __restrict__ w1t,   // [128,128]
                                        const float*  __restrict__ b1,
                                        const float*  __restrict__ vhull, // [N,128]
                                        const int*    __restrict__ jrow,  // [EH]
                                        const int*    __restrict__ irow,  // [EH]
                                        float* __restrict__ out_hull) {   // [N,128]
  __shared__ __bf16 hid_s[8][16 * 128];   // 4 KB per wave, 8 waves/block
  int wslot = threadIdx.x >> 5;
  int lane  = threadIdx.x & 31;
  int tile  = blockIdx.x * (blockDim.x >> 5) + wslot;
  if ((tile << 4) >= E_HULL) return;
  int e0 = tile << 4;
  __bf16* hid = hid_s[wslot];

  // ---- layer 1: [16,32] x [32,128], silu, -> LDS bf16 -------------------
  v16bf a0 = load_frag(feaP, 32, e0, 0, lane);
  int rbase = (lane & 16) >> 1;
  int nl = lane & 15;
#pragma unroll
  for (int nt = 0; nt < 8; ++nt) {
    v16bf b = load_frag(w0t, 32, nt << 4, 0, lane);
    v8f acc = {0.f, 0.f, 0.f, 0.f, 0.f, 0.f, 0.f, 0.f};
    acc = __builtin_amdgcn_wmma_f32_16x16x32_bf16(false, a0, false, b,
                                                  (short)0, acc, false, false);
    int n = (nt << 4) + nl;
    float bb = b0[n];
#pragma unroll
    for (int r = 0; r < 8; ++r)
      hid[(rbase + r) * 128 + n] = f2bf(silu_f(acc[r] + bb));
  }

  // ---- layer 2: [16,128] x [128,128] -> 8 accumulators ------------------
  v8f accs[8];
#pragma unroll
  for (int nt = 0; nt < 8; ++nt) {
    v8f z = {0.f, 0.f, 0.f, 0.f, 0.f, 0.f, 0.f, 0.f};
    accs[nt] = z;
  }
  for (int k0 = 0; k0 < 128; k0 += 32) {
    v16bf a = load_frag(hid, 128, 0, k0, lane);
#pragma unroll
    for (int nt = 0; nt < 8; ++nt) {
      v16bf b = load_frag(w1t, 128, nt << 4, k0, lane);
      accs[nt] = __builtin_amdgcn_wmma_f32_16x16x32_bf16(false, a, false, b,
                                                         (short)0, accs[nt], false, false);
    }
  }

  // ---- gather v_hull[j], multiply, atomic scatter to out_hull[i] --------
  int jj[8], ii[8];
#pragma unroll
  for (int r = 0; r < 8; ++r) {
    int e = e0 + rbase + r;
    jj[r] = jrow[e];
    ii[r] = irow[e];
  }
#pragma unroll
  for (int nt = 0; nt < 8; ++nt) {
    int n = (nt << 4) + nl;
    float bb = b1[n];
#pragma unroll
    for (int r = 0; r < 8; ++r) {
      float wv  = accs[nt][r] + bb;
      float val = wv * vhull[(size_t)jj[r] * 128 + n];
      atomicAdd(out_hull + (size_t)ii[r] * 128 + n, val);
    }
  }
}

// final projection: out[n] = sum_k r[n,k] * w_out[k]   (K=256, OC=1)
// One wave per row: each lane loads 8 contiguous bf16 (one B128), then
// a 5-step shuffle-xor reduction across the wave.
__global__ void out_dot_kernel(const __bf16* __restrict__ r,
                               const float* __restrict__ wout,
                               float* __restrict__ out) {
  int wave = blockIdx.x * (blockDim.x >> 5) + (threadIdx.x >> 5);
  int lane = threadIdx.x & 31;
  if (wave >= N_NODES) return;
  const v8bf* p = (const v8bf*)(r + (size_t)wave * 256) + lane;
  v8bf d = *p;
  const float* wv = wout + lane * 8;
  float s = 0.0f;
#pragma unroll
  for (int k = 0; k < 8; ++k) s += (float)d[k] * wv[k];
#pragma unroll
  for (int m = 16; m > 0; m >>= 1) s += __shfl_xor(s, m, 32);
  if (lane == 0) out[wave] = s;
}

// ---------------------------------------------------------------------------
// Host side
// ---------------------------------------------------------------------------
static inline void launch_gemm(hipStream_t s, const __bf16* A, int lda,
                               const __bf16* Wt, const float* bias,
                               int M, int K, int Nout,
                               float* Cf, __bf16* Cb, int ldc, int coloff, int dosilu) {
  int waves  = (M / 16) * (Nout / 64);        // one wave per 16x64 tile
  int blocks = (waves + 7) / 8;               // 8 waves (256 threads) per block
  gemm_bf16_kernel<<<blocks, 256, 0, s>>>(A, lda, Wt, bias, M, K, Nout,
                                          Cf, Cb, ldc, coloff, dosilu);
}

extern "C" void kernel_launch(void* const* d_in, const int* in_sizes, int n_in,
                              void* d_out, int out_size, void* d_ws, size_t ws_size,
                              hipStream_t stream) {
  (void)in_sizes; (void)n_in; (void)out_size; (void)ws_size;

  const float* e2     = (const float*)d_in[0];
  const int*   idx    = (const int*)d_in[1];
  const float* fea    = (const float*)d_in[2];
  const int*   eih    = (const int*)d_in[3];
  const int*   jrow   = eih;            // edge_index_hull[0]
  const int*   irow   = eih + E_HULL;   // edge_index_hull[1]
  const float* w_hull = (const float*)d_in[4];
  const float* w_mlp0 = (const float*)d_in[5];
  const float* b_mlp0 = (const float*)d_in[6];
  const float* w_mlp1 = (const float*)d_in[7];
  const float* b_mlp1 = (const float*)d_in[8];
  const float* w1h    = (const float*)d_in[9];
  const float* b1h    = (const float*)d_in[10];
  const float* w2h    = (const float*)d_in[11];
  const float* b2h    = (const float*)d_in[12];
  const float* w_cat  = (const float*)d_in[13];
  const float* b_cat  = (const float*)d_in[14];
  const float* w_up   = (const float*)d_in[15];
  const float* b_up   = (const float*)d_in[16];
  const float* w_l0   = (const float*)d_in[17];
  const float* b_l0   = (const float*)d_in[18];
  const float* w_l1   = (const float*)d_in[19];
  const float* b_l1   = (const float*)d_in[20];
  const float* w_out  = (const float*)d_in[21];
  float* out = (float*)d_out;

  // ---- workspace carve (with lifetime-based aliasing) -------------------
  size_t off = 0;
  char* ws = (char*)d_ws;
  auto alloc = [&](size_t bytes) -> void* {
    void* p = ws + off;
    off += (bytes + 255) & ~(size_t)255;
    return p;
  };
  const size_t N = N_NODES;
  float*  v_f32    = (float*)alloc(N * 128 * 4);        // later reused as p_bf
  float*  vhull    = (float*)alloc(N * 128 * 4);        // later reused as u_bf
  float*  out_hull = (float*)alloc(N * 128 * 4);        // later reused as q_bf
  __bf16* cat_bf   = (__bf16*)alloc(N * 384 * 2);       // later reused as r_bf
  __bf16* fea_pad  = (__bf16*)alloc((size_t)E_HULL * 32 * 2); // later reused as oh_bf
  __bf16* h1_bf    = (__bf16*)alloc(N * 128 * 2);
  __bf16* w_hullT  = (__bf16*)alloc(128 * 128 * 2);
  __bf16* mlp0T    = (__bf16*)alloc(128 * 32 * 2);
  __bf16* mlp1T    = (__bf16*)alloc(128 * 128 * 2);
  __bf16* w1T      = (__bf16*)alloc(128 * 128 * 2);
  __bf16* w2T      = (__bf16*)alloc(256 * 128 * 2);
  __bf16* wcatT    = (__bf16*)alloc(128 * 384 * 2);
  __bf16* wupT     = (__bf16*)alloc(256 * 128 * 2);
  __bf16* wl0T     = (__bf16*)alloc(256 * 256 * 2);
  __bf16* wl1T     = (__bf16*)alloc(256 * 256 * 2);
  // aliases (disjoint lifetimes)
  __bf16* oh_bf = fea_pad;         // out_hull in bf16 (fea_pad dead by then)
  __bf16* u_bf  = (__bf16*)vhull;  // after edge kernel consumed vhull
  __bf16* p_bf  = (__bf16*)v_f32;  // after v converted into cat
  __bf16* q_bf  = (__bf16*)out_hull;
  __bf16* r_bf  = cat_bf;          // after w_cat GEMM consumed cat

  // ---- weight / feature preparation -------------------------------------
  auto gpw = [&](const float* w, __bf16* wt, int in, int outd, int pad) {
    size_t n = (size_t)outd * pad;
    prep_wt_kernel<<<(int)((n + 255) / 256), 256, 0, stream>>>(w, wt, in, outd, pad);
  };
  gpw(w_hull, w_hullT, 128, 128, 128);
  gpw(w_mlp0, mlp0T,   16,  128, 32);
  gpw(w_mlp1, mlp1T,   128, 128, 128);
  gpw(w1h,    w1T,     128, 128, 128);
  gpw(w2h,    w2T,     128, 256, 128);
  gpw(w_cat,  wcatT,   384, 128, 384);
  gpw(w_up,   wupT,    128, 256, 128);
  gpw(w_l0,   wl0T,    256, 256, 256);
  gpw(w_l1,   wl1T,    256, 256, 256);
  {
    size_t n = (size_t)E_HULL * 32;
    prep_fea_kernel<<<(int)((n + 255) / 256), 256, 0, stream>>>(fea, fea_pad);
  }

  // ---- 1) v = segment_sum(e2, i) ----------------------------------------
  {
    size_t n = N * 128;
    zero_f32<<<(int)((n + 255) / 256), 256, 0, stream>>>(v_f32, n);
    zero_f32<<<(int)((n + 255) / 256), 256, 0, stream>>>(out_hull, n);
    size_t tot = (size_t)E_MAIN * 32;   // one thread per edge x 4 channels
    scatter_v_kernel<<<(int)((tot + 255) / 256), 256, 0, stream>>>(
        (const float4*)e2, idx, v_f32);
  }

  // ---- 2) v -> bf16 into cat[:, 0:128] ----------------------------------
  {
    size_t n = N * 128;
    f32_to_bf16_kernel<<<(int)((n + 255) / 256), 256, 0, stream>>>(
        v_f32, cat_bf, (int)N, 128, 384, 0);
  }

  // ---- 3) v_hull = v @ w_hull  (f32 out, K=128) -------------------------
  launch_gemm(stream, cat_bf, 384, w_hullT, nullptr, (int)N, 128, 128,
              vhull, nullptr, 128, 0, 0);

  // ---- 4) fused edge MLP + gather-mul + scatter-sum ---------------------
  edge_mlp_scatter_kernel<<<E_HULL / 16 / 8, 256, 0, stream>>>(
      fea_pad, mlp0T, b_mlp0, mlp1T, b_mlp1, vhull, jrow, irow, out_hull);

  // ---- 5) out_hull -> bf16 ----------------------------------------------
  {
    size_t n = N * 128;
    f32_to_bf16_kernel<<<(int)((n + 255) / 256), 256, 0, stream>>>(
        out_hull, oh_bf, (int)N, 128, 128, 0);
  }

  // ---- 6) h1 = silu(out_hull @ w1_hull + b1) ----------------------------
  launch_gemm(stream, oh_bf, 128, w1T, b1h, (int)N, 128, 128,
              nullptr, h1_bf, 128, 0, 1);

  // ---- 7) cat[:, 128:384] = h1 @ w2_hull + b2 ---------------------------
  launch_gemm(stream, h1_bf, 128, w2T, b2h, (int)N, 128, 256,
              nullptr, cat_bf, 384, 128, 0);

  // ---- 8) u = silu(cat @ w_cat + b_cat)  (K=384) ------------------------
  launch_gemm(stream, cat_bf, 384, wcatT, b_cat, (int)N, 384, 128,
              nullptr, u_bf, 128, 0, 1);

  // ---- 9) p = u @ w_up + b_up -------------------------------------------
  launch_gemm(stream, u_bf, 128, wupT, b_up, (int)N, 128, 256,
              nullptr, p_bf, 256, 0, 0);

  // ---- 10) q = silu(p @ w_l0 + b_l0) ------------------------------------
  launch_gemm(stream, p_bf, 256, wl0T, b_l0, (int)N, 256, 256,
              nullptr, q_bf, 256, 0, 1);

  // ---- 11) r = silu(q @ w_l1 + b_l1) ------------------------------------
  launch_gemm(stream, q_bf, 256, wl1T, b_l1, (int)N, 256, 256,
              nullptr, r_bf, 256, 0, 1);

  // ---- 12) out = r @ w_out ----------------------------------------------
  out_dot_kernel<<<(N_NODES + 7) / 8, 256, 0, stream>>>(r_bf, w_out, out);
}
